// GATModel_31001073943306
// MI455X (gfx1250) — compile-verified, matrix operand order checked by reading, not run on hardware
//
#include <hip/hip_runtime.h>
#include <hip/hip_bf16.h>
#include <math.h>

#define Nn 50000
#define Ee 800000
#define INF_ 384
#define Hh 8
#define HDd 32
#define HIDd 256
#define OUTd 192
#define Cc 400
#define Bb 16
#define SLOPE 0.2f

typedef __attribute__((ext_vector_type(16))) _Float16 v16h;
typedef __attribute__((ext_vector_type(8)))  _Float16 v8h;
typedef __attribute__((ext_vector_type(8)))  float    v8f;

// ---------------- helpers ----------------

__device__ __forceinline__ float lrelu(float z) { return z > 0.0f ? z : SLOPE * z; }

__device__ __forceinline__ void atomicMaxF32(float* addr, float val) {
    // signed-int max for non-negative floats, unsigned-int min for negatives.
    if (val >= 0.0f) {
        atomicMax((int*)addr, __float_as_int(val));
    } else {
        atomicMin((unsigned int*)addr, __float_as_uint(val));
    }
}

__device__ __forceinline__ float warpSum32(float v) {
#pragma unroll
    for (int off = 16; off > 0; off >>= 1) v += __shfl_xor(v, off, 32);
    return v;
}

// ---------------- prep kernels ----------------

__global__ void fill_f32(float* p, float v, int n) {
    int i = blockIdx.x * blockDim.x + threadIdx.x;
    if (i < n) p[i] = v;
}

__global__ void cvt_f32_to_f16(const float* __restrict__ a, _Float16* __restrict__ o, int n) {
    int i = blockIdx.x * blockDim.x + threadIdx.x;
    if (i < n) o[i] = (_Float16)a[i];
}

// Pack B [K,Ncols] f32 row-major into per-lane WMMA B-fragment order:
//   Bp[(((k0/32)*Ncols + n)*2 + half)*16 + p], p=0..15 maps to (v=p/2, j=p%2),
//   kb = (v/4)*16 + half*8 + (v%4)*2 + j, element = B[(k0+kb)*Ncols + n].
__global__ void pack_B_f16(const float* __restrict__ B, _Float16* __restrict__ Bp,
                           int K, int Ncols) {
    int i = blockIdx.x * blockDim.x + threadIdx.x;
    if (i >= K * Ncols) return;
    int p  = i & 15;
    int hf = (i >> 4) & 1;
    int n  = (i >> 5) % Ncols;
    int t  = (i >> 5) / Ncols;
    int v = p >> 1, j = p & 1;
    int kb = (v >> 2) * 16 + hf * 8 + (v & 3) * 2 + j;
    int k = t * 32 + kb;
    Bp[i] = (_Float16)B[(size_t)k * Ncols + n];
}

// ---------------- WMMA GEMM (f16 operands pre-packed, f32 accum/out) ----------------
// A: [M,K] f16 row-major; Bp: fragment-packed f16; C: [M,NCOLS] f32.
// One wave -> 16 x 64 strip of C. grid = (M/16, NCOLS/64). K % 32 == 0, NCOLS % 64 == 0.

template <int K, int NCOLS>
__global__ __launch_bounds__(32) void gemm_wmma_f16p(const _Float16* __restrict__ A,
                                                     const _Float16* __restrict__ Bp,
                                                     float* __restrict__ C) {
    int lane = threadIdx.x & 31;
    int half = lane >> 4;
    int nlo = lane & 15;
    int m0 = blockIdx.x * 16;
    int n0 = blockIdx.y * 64;

    const _Float16* arow = A + (size_t)(m0 + nlo) * K + half * 8;
    const _Float16* bbase = Bp + ((size_t)(n0 + nlo) * 2 + half) * 16;

    v8f acc[4] = {{}, {}, {}, {}};
    for (int k0 = 0; k0 < K; k0 += 32) {
        v8h alo = *(const v8h*)(arow + k0);        // K = half*8 .. +7   (VGPR 0..3)
        v8h ahi = *(const v8h*)(arow + k0 + 16);   // K = 16+half*8 ..+7 (VGPR 4..7)
        v16h a;
#pragma unroll
        for (int i = 0; i < 8; ++i) { a[i] = alo[i]; a[8 + i] = ahi[i]; }
        if (k0 + 32 < K) __builtin_prefetch(arow + k0 + 32, 0, 3);

        const _Float16* bt = bbase + (size_t)(k0 >> 5) * NCOLS * 32;
#pragma unroll
        for (int c = 0; c < 4; ++c) {
            v16h b = *(const v16h*)(bt + (size_t)c * 16 * 32);
            acc[c] = __builtin_amdgcn_wmma_f32_16x16x32_f16(false, a, false, b,
                                                            (short)0, acc[c], false, false);
        }
    }

#pragma unroll
    for (int c = 0; c < 4; ++c) {
        int n = n0 + c * 16 + nlo;
#pragma unroll
        for (int v = 0; v < 8; ++v) {
            int m = m0 + v + 8 * half;
            C[(size_t)m * NCOLS + n] = acc[c][v];
        }
    }
}

// ---------------- GAT layer 1 edge kernels (H=8, HD=32) ----------------

__global__ void edge_logits1(const float* __restrict__ h1, const int* __restrict__ src,
                             const int* __restrict__ dst, const float* __restrict__ a1,
                             float* __restrict__ elog, float* __restrict__ nmax) {
    int t = blockIdx.x * blockDim.x + threadIdx.x;
    if (t >= Ee * Hh) return;
    int e = t >> 3, h = t & 7;
    int s = src[e], d = dst[e];
    const float* hs = h1 + (size_t)s * HIDd + h * HDd;
    const float* hd = h1 + (size_t)d * HIDd + h * HDd;
    const float* ah = a1 + h * HDd;
    float acc = 0.0f;
#pragma unroll
    for (int i = 0; i < HDd; ++i) {
        acc += ah[i] * lrelu(hs[i] + hd[i]);
    }
    elog[t] = acc;
    atomicMaxF32(nmax + (size_t)d * Hh + h, acc);
}

__global__ void edge_expsum1(const int* __restrict__ dst, float* __restrict__ elog,
                             const float* __restrict__ nmax, float* __restrict__ denom) {
    int t = blockIdx.x * blockDim.x + threadIdx.x;
    if (t >= Ee * Hh) return;
    int e = t >> 3, h = t & 7;
    int d = dst[e];
    float v = expf(elog[t] - nmax[(size_t)d * Hh + h]);
    elog[t] = v;
    atomicAdd(denom + (size_t)d * Hh + h, v);
}

__global__ void edge_scatter1(const float* __restrict__ h1, const int* __restrict__ src,
                              const int* __restrict__ dst, const float* __restrict__ elog,
                              const float* __restrict__ denom, float* __restrict__ out1) {
    int t = blockIdx.x * blockDim.x + threadIdx.x;
    if (t >= Ee * Hh) return;
    int e = t >> 3, h = t & 7;
    int s = src[e], d = dst[e];
    float alpha = elog[t] / (denom[(size_t)d * Hh + h] + 1e-16f);
    const float* hs = h1 + (size_t)s * HIDd + h * HDd;
    float* o = out1 + (size_t)d * HIDd + h * HDd;
#pragma unroll
    for (int i = 0; i < HDd; ++i) {
        atomicAdd(o + i, alpha * hs[i]);
    }
}

// ---------------- layernorm + leaky relu, one wave per node ----------------

// In-place f32 variant (final LN).
template <int D>
__global__ void ln_lrelu(float* __restrict__ x, const float* __restrict__ g,
                         const float* __restrict__ b, int n) {
    int wid = (blockIdx.x * blockDim.x + threadIdx.x) >> 5;
    int lane = threadIdx.x & 31;
    if (wid >= n) return;
    const int per = D / 32;
    float* row = x + (size_t)wid * D;
    float vals[8];
    float s = 0.0f, s2 = 0.0f;
#pragma unroll
    for (int i = 0; i < per; ++i) {
        float v = row[lane * per + i];
        vals[i] = v;
        s += v;
        s2 += v * v;
    }
    s = warpSum32(s);
    s2 = warpSum32(s2);
    float m = s / (float)D;
    float var = s2 / (float)D - m * m;
    float inv = rsqrtf(var + 1e-5f);
#pragma unroll
    for (int i = 0; i < per; ++i) {
        int c = lane * per + i;
        float y = (vals[i] - m) * inv * g[c] + b[c];
        row[c] = lrelu(y);
    }
}

// f32 in -> f16 out variant (feeds GEMM2 directly).
template <int D>
__global__ void ln_lrelu_f16out(const float* __restrict__ x, _Float16* __restrict__ o,
                                const float* __restrict__ g, const float* __restrict__ b,
                                int n) {
    int wid = (blockIdx.x * blockDim.x + threadIdx.x) >> 5;
    int lane = threadIdx.x & 31;
    if (wid >= n) return;
    const int per = D / 32;
    const float* row = x + (size_t)wid * D;
    _Float16* orow = o + (size_t)wid * D;
    float vals[8];
    float s = 0.0f, s2 = 0.0f;
#pragma unroll
    for (int i = 0; i < per; ++i) {
        float v = row[lane * per + i];
        vals[i] = v;
        s += v;
        s2 += v * v;
    }
    s = warpSum32(s);
    s2 = warpSum32(s2);
    float m = s / (float)D;
    float var = s2 / (float)D - m * m;
    float inv = rsqrtf(var + 1e-5f);
#pragma unroll
    for (int i = 0; i < per; ++i) {
        int c = lane * per + i;
        float y = (vals[i] - m) * inv * g[c] + b[c];
        orow[c] = (_Float16)lrelu(y);
    }
}

// ---------------- GAT layer 2 edge kernels (1 head, 192-wide), warp per edge ----------------

__global__ void edge_logits2(const float* __restrict__ h2, const int* __restrict__ src,
                             const int* __restrict__ dst, const float* __restrict__ a2,
                             float* __restrict__ e2, float* __restrict__ nmax2) {
    int w = (blockIdx.x * blockDim.x + threadIdx.x) >> 5;
    int lane = threadIdx.x & 31;
    if (w >= Ee) return;
    int s = src[w], d = dst[w];
    const float* hs = h2 + (size_t)s * OUTd + lane * 6;
    const float* hd = h2 + (size_t)d * OUTd + lane * 6;
    const float* a = a2 + lane * 6;
    float acc = 0.0f;
#pragma unroll
    for (int i = 0; i < 6; ++i) {
        acc += a[i] * lrelu(hs[i] + hd[i]);
    }
    acc = warpSum32(acc);
    if (lane == 0) {
        e2[w] = acc;
        atomicMaxF32(nmax2 + d, acc);
    }
}

__global__ void edge_expsum2(const int* __restrict__ dst, float* __restrict__ e2,
                             const float* __restrict__ nmax2, float* __restrict__ denom2) {
    int e = blockIdx.x * blockDim.x + threadIdx.x;
    if (e >= Ee) return;
    int d = dst[e];
    float v = expf(e2[e] - nmax2[d]);
    e2[e] = v;
    atomicAdd(denom2 + d, v);
}

__global__ void edge_scatter2(const float* __restrict__ h2, const int* __restrict__ src,
                              const int* __restrict__ dst, const float* __restrict__ e2,
                              const float* __restrict__ denom2, float* __restrict__ out2) {
    int w = (blockIdx.x * blockDim.x + threadIdx.x) >> 5;
    int lane = threadIdx.x & 31;
    if (w >= Ee) return;
    int s = src[w], d = dst[w];
    float alpha = e2[w] / (denom2[d] + 1e-16f);
    const float* hs = h2 + (size_t)s * OUTd + lane * 6;
    float* o = out2 + (size_t)d * OUTd + lane * 6;
#pragma unroll
    for (int i = 0; i < 6; ++i) {
        atomicAdd(o + i, alpha * hs[i]);
    }
}

// ---------------- classifier: out[b,c] = hf[seg[b]] . Wc[:,c] + bc[c] ----------------

__global__ void classify(const float* __restrict__ hf, const int* __restrict__ seg,
                         const float* __restrict__ Wc, const float* __restrict__ bc,
                         float* __restrict__ out) {
    int bi = blockIdx.x;
    int node = seg[bi];
    const float* row = hf + (size_t)node * OUTd;
    for (int c = threadIdx.x; c < Cc; c += blockDim.x) {
        float acc = bc[c];
#pragma unroll 4
        for (int d = 0; d < OUTd; ++d) {
            acc += row[d] * Wc[(size_t)d * Cc + c];
        }
        out[(size_t)bi * Cc + c] = acc;
    }
}

// ---------------- launch ----------------

extern "C" void kernel_launch(void* const* d_in, const int* in_sizes, int n_in,
                              void* d_out, int out_size, void* d_ws, size_t ws_size,
                              hipStream_t stream) {
    const float* x   = (const float*)d_in[0];
    const int*   adj = (const int*)d_in[1];
    const int*   seg = (const int*)d_in[2];
    const float* W1  = (const float*)d_in[3];
    const float* a1  = (const float*)d_in[4];
    const float* W2  = (const float*)d_in[5];
    const float* a2  = (const float*)d_in[6];
    const float* g1  = (const float*)d_in[7];
    const float* b1  = (const float*)d_in[8];
    const float* g2  = (const float*)d_in[9];
    const float* b2  = (const float*)d_in[10];
    const float* Wc  = (const float*)d_in[11];
    const float* bc  = (const float*)d_in[12];
    float* out = (float*)d_out;

    const int* src = adj;          // adj[0, :]
    const int* dst = adj + Ee;     // adj[1, :]

    // workspace layout (256B aligned).
    size_t off = 0;
    auto take = [&](size_t bytes) { size_t o = off; off += (bytes + 255) & ~(size_t)255; return o; };
    _Float16* xh    = (_Float16*)((char*)d_ws + take((size_t)Nn * INF_ * 2));  // x in f16
    _Float16* W1p   = (_Float16*)((char*)d_ws + take((size_t)INF_ * HIDd * 2)); // packed W1
    _Float16* W2p   = (_Float16*)((char*)d_ws + take((size_t)HIDd * OUTd * 2)); // packed W2
    _Float16* h1nh  = (_Float16*)((char*)d_ws + take((size_t)Nn * HIDd * 2));  // post-LN1, f16
    float* h1    = (float*)((char*)d_ws + take((size_t)Nn * HIDd * 4)); // [N,256]; reused as h2 [N,192]
    float* out1  = (float*)((char*)d_ws + take((size_t)Nn * HIDd * 4)); // [N,256] gat1 output
    float* out2  = (float*)((char*)d_ws + take((size_t)Nn * OUTd * 4)); // [N,192] gat2 output, LN in place
    float* elog  = (float*)((char*)d_ws + take((size_t)Ee * Hh * 4));   // [E,8]; reused as e2 [E]
    float* nmax  = (float*)((char*)d_ws + take((size_t)Nn * Hh * 4));   // [N,8]; reused as nmax2 [N]
    float* denom = (float*)((char*)d_ws + take((size_t)Nn * Hh * 4));   // [N,8]; reused as denom2 [N]
    (void)ws_size;

    float* h2     = h1;   // h1 dead after edge_scatter1
    float* e2     = elog;
    float* nmax2  = nmax;
    float* denom2 = denom;

    // --- prep: conversions / packing / inits ---
    cvt_f32_to_f16<<<(Nn * INF_ + 255) / 256, 256, 0, stream>>>(x, xh, Nn * INF_);
    pack_B_f16<<<(INF_ * HIDd + 255) / 256, 256, 0, stream>>>(W1, W1p, INF_, HIDd);
    pack_B_f16<<<(HIDd * OUTd + 255) / 256, 256, 0, stream>>>(W2, W2p, HIDd, OUTd);
    fill_f32<<<(Nn * Hh + 255) / 256, 256, 0, stream>>>(nmax, -INFINITY, Nn * Hh);
    hipMemsetAsync(denom, 0, (size_t)Nn * Hh * 4, stream);
    hipMemsetAsync(out1, 0, (size_t)Nn * HIDd * 4, stream);

    // --- h1 = x @ W1  [50000,384]x[384,256] ---
    gemm_wmma_f16p<INF_, HIDd><<<dim3(Nn / 16, HIDd / 64), 32, 0, stream>>>(xh, W1p, h1);

    // --- gat1 edge softmax + scatter ---
    int eh_blocks = (Ee * Hh + 255) / 256;
    edge_logits1<<<eh_blocks, 256, 0, stream>>>(h1, src, dst, a1, elog, nmax);
    edge_expsum1<<<eh_blocks, 256, 0, stream>>>(dst, elog, nmax, denom);
    edge_scatter1<<<eh_blocks, 256, 0, stream>>>(h1, src, dst, elog, denom, out1);

    // --- LN1 + leaky relu -> f16 activations for GEMM2 ---
    ln_lrelu_f16out<HIDd><<<(Nn + 7) / 8, 256, 0, stream>>>(out1, h1nh, g1, b1, Nn);

    // --- h2 = ln1(out1) @ W2  [50000,256]x[256,192] (overwrites h1 buffer) ---
    gemm_wmma_f16p<HIDd, OUTd><<<dim3(Nn / 16, OUTd / 64), 32, 0, stream>>>(h1nh, W2p, h2);

    // --- gat2 inits ---
    fill_f32<<<(Nn + 255) / 256, 256, 0, stream>>>(nmax2, -INFINITY, Nn);
    hipMemsetAsync(denom2, 0, (size_t)Nn * 4, stream);
    hipMemsetAsync(out2, 0, (size_t)Nn * OUTd * 4, stream);

    // --- gat2 edge softmax + scatter (warp per edge) ---
    int ew_blocks = (Ee * 32 + 255) / 256;
    edge_logits2<<<ew_blocks, 256, 0, stream>>>(h2, src, dst, a2, e2, nmax2);
    edge_expsum2<<<(Ee + 255) / 256, 256, 0, stream>>>(dst, e2, nmax2, denom2);
    edge_scatter2<<<ew_blocks, 256, 0, stream>>>(h2, src, dst, e2, denom2, out2);

    // --- final LN + leaky relu (in place on out2) ---
    ln_lrelu<OUTd><<<(Nn + 7) / 8, 256, 0, stream>>>(out2, g2, b2, Nn);

    // --- classifier ---
    classify<<<Bb, 256, 0, stream>>>(out2, seg, Wc, bc, out);
}
